// Note_Decoder_43387759624351
// MI455X (gfx1250) — compile-verified, matrix operand order checked by reading, not run on hardware
//
#include <hip/hip_runtime.h>
#include <cstdint>
#include <cstddef>

typedef _Float16 v16h __attribute__((ext_vector_type(16)));
typedef _Float16 v8h  __attribute__((ext_vector_type(8)));
typedef float    v8f  __attribute__((ext_vector_type(8)));

#define B_TOT 512
#define CD    64
#define ZL    16
#define HID   512
#define H2C   256
#define NC    9
#define SLEN  64
#define K1PAD 288   // 274 (9+256+9) padded up to a multiple of 32

// ---------------- WMMA helpers (CDNA5 16x16x32 f16, fp32 accum) ----------------
// A (16x32 f16): lane<16 -> row M=lane, halves = K{0..7,16..23}; lane>=16 -> M=lane-16, K{8..15,24..31}
// B (32x16 f16) loaded from row-major W[N][K] with the symmetric layout (N in lane, same K split).
// C/D (16x16 f32): elem e -> M = e + 8*(lane>=16), N = lane&15.
__device__ __forceinline__ v16h ld_frag(const _Float16* p) {
  v16h a;
  ((v8h*)&a)[0] = *(const v8h*)(p);       // K + {0..7}
  ((v8h*)&a)[1] = *(const v8h*)(p + 16);  // K + {16..23}
  return a;
}
__device__ __forceinline__ v8f wmma_f16(v16h a, v16h b, v8f c) {
  return __builtin_amdgcn_wmma_f32_16x16x32_f16(false, a, false, b, (short)0, c, false, false);
}
__device__ __forceinline__ float sigm(float x) { return 1.f / (1.f + expf(-x)); }

// ---------------- threefry2x32 (JAX PRNG replication) ----------------
__device__ __forceinline__ void tf2x32(unsigned k0, unsigned k1, unsigned c0, unsigned c1,
                                       unsigned& o0, unsigned& o1) {
  unsigned ks[3] = {k0, k1, k0 ^ k1 ^ 0x1BD11BDAu};
  unsigned x0 = c0 + k0, x1 = c1 + k1;
  const unsigned rot[2][4] = {{13u,15u,26u,6u},{17u,29u,16u,24u}};
#pragma unroll
  for (int g = 0; g < 5; ++g) {
    const unsigned* r = rot[g & 1];
#pragma unroll
    for (int i = 0; i < 4; ++i) { x0 += x1; x1 = (x1 << r[i]) | (x1 >> (32u - r[i])); x1 ^= x0; }
    x0 += ks[(g + 1) % 3];
    x1 += ks[(g + 2) % 3] + (unsigned)(g + 1);
  }
  o0 = x0; o1 = x1;
}

// ---------------- fp32 -> fp16 weight conversion with row + K padding ----------------
__global__ __launch_bounds__(256)
void cvt_pad_half(const float* __restrict__ src, _Float16* __restrict__ dst,
                  int rsrc, int rdst, int ksrc, int kdst) {
  int i = blockIdx.x * 256 + threadIdx.x;
  int total = rdst * kdst;
  if (i < total) {
    int r = i / kdst, k = i - r * kdst;
    dst[i] = (r < rsrc && k < ksrc) ? (_Float16)src[(size_t)r * ksrc + k] : (_Float16)0.f;
  }
}

// ---------------- ConvTranspose1d stage 1: z[B,64,16] -> y1[B,32,512] (t-major) ----------------
// stride 2, k=4: output t uses exactly 2 taps selected by parity. Includes bias+BN+LeakyReLU(0.2).
__global__ __launch_bounds__(512)
void convt1_kernel(const float* __restrict__ z, const float* __restrict__ w,
                   const float* __restrict__ bias,
                   const float* __restrict__ g, const float* __restrict__ bt,
                   const float* __restrict__ mu, const float* __restrict__ var,
                   float* __restrict__ y1) {
  const int blk = blockIdx.x;
  const int b = blk >> 5;
  const int t = blk & 31;
  const int c = threadIdx.x;
  __shared__ float zA[CD], zB[CD];
  int iA, iB, tapA, tapB;
  if ((t & 1) == 0) { iA = (t - 2) >> 1; tapA = 2; iB = t >> 1;       tapB = 0; }
  else              { iA = (t - 3) >> 1; tapA = 3; iB = (t - 1) >> 1; tapB = 1; }
  if (c < CD)        zA[c]       = (iA >= 0 && iA < ZL) ? z[((size_t)b * CD + c) * ZL + iA] : 0.f;
  else if (c < 2*CD) zB[c - CD]  = (iB >= 0 && iB < ZL) ? z[((size_t)b * CD + (c - CD)) * ZL + iB] : 0.f;
  __syncthreads();
  float acc = bias[c];
#pragma unroll 4
  for (int ci = 0; ci < CD; ++ci)
    acc += zA[ci] * w[(ci * HID + c) * 4 + tapA] + zB[ci] * w[(ci * HID + c) * 4 + tapB];
  float yn = (acc - mu[c]) * (g[c] * rsqrtf(var[c] + 1e-5f)) + bt[c];
  y1[((size_t)b * 32 + t) * HID + c] = (yn > 0.f) ? yn : 0.2f * yn;
}

// ---------------- ConvTranspose1d stage 2: y1 -> inputs[B,64,256] fp16 ----------------
__global__ __launch_bounds__(256)
void convt2_kernel(const float* __restrict__ y1, const float* __restrict__ w,
                   const float* __restrict__ bias,
                   const float* __restrict__ g, const float* __restrict__ bt,
                   const float* __restrict__ mu, const float* __restrict__ var,
                   _Float16* __restrict__ inph) {
  const int blk = blockIdx.x;
  const int b = blk >> 6;
  const int t = blk & 63;
  const int c = threadIdx.x; // output channel 0..255
  __shared__ float xA[HID], xB[HID];
  int iA, iB, tapA, tapB;
  if ((t & 1) == 0) { iA = (t - 2) >> 1; tapA = 2; iB = t >> 1;       tapB = 0; }
  else              { iA = (t - 3) >> 1; tapA = 3; iB = (t - 1) >> 1; tapB = 1; }
#pragma unroll
  for (int q = 0; q < 2; ++q) {
    int ci = c + q * 256;
    xA[ci] = (iA >= 0 && iA < 32) ? y1[((size_t)b * 32 + iA) * HID + ci] : 0.f;
    xB[ci] = (iB >= 0 && iB < 32) ? y1[((size_t)b * 32 + iB) * HID + ci] : 0.f;
  }
  __syncthreads();
  float acc = bias[c];
#pragma unroll 4
  for (int ci = 0; ci < HID; ++ci)
    acc += xA[ci] * w[(ci * H2C + c) * 4 + tapA] + xB[ci] * w[(ci * H2C + c) * 4 + tapB];
  float yn = (acc - mu[c]) * (g[c] * rsqrtf(var[c] + 1e-5f)) + bt[c];
  yn = (yn > 0.f) ? yn : 0.2f * yn;
  inph[((size_t)b * SLEN + t) * H2C + c] = (_Float16)yn;
}

// ---------------- GRU cell core: two-pass WMMA GEMM (r/z then n), in-place state update ----------
// Wave owns 64 hidden columns (4 N-tiles), identical columns across the r/z/n gate blocks.
// Pass split keeps only 64 accumulator VGPRs live at a time (r,z parked in LDS) -> no spills.
__device__ __forceinline__ void gru_core(
    const _Float16* __restrict__ A1, int K1t,             // input activations (LDS), row stride K1t*32
    _Float16* __restrict__ hB, float* __restrict__ hF,    // state: fp16 GEMM copy + fp32 master (LDS)
    float* __restrict__ rz,                               // LDS scratch: r at [0], z at [16*HID]
    const _Float16* __restrict__ Wih, int Kih,            // [1536][Kih] fp16 (global)
    const _Float16* __restrict__ Whh,                     // [1536][512] fp16 (global)
    const float* __restrict__ bih, const float* __restrict__ bhh,
    int m, int kh, int ncol0) {
  const int ld1 = K1t * 32;

  // ---- pass A: r and z gates ----
  {
    v8f aR[4] = { {}, {}, {}, {} };
    v8f aZ[4] = { {}, {}, {}, {} };
    for (int kt = 0; kt < K1t; ++kt) {
      v16h a = ld_frag(A1 + m * ld1 + kt * 32 + kh * 8);
      const int kof = kt * 32 + kh * 8;
#pragma unroll
      for (int j = 0; j < 4; ++j) {
        const size_t nr = (size_t)(ncol0 + j * 16 + m);
        aR[j] = wmma_f16(a, ld_frag(Wih + nr * Kih + kof), aR[j]);
        aZ[j] = wmma_f16(a, ld_frag(Wih + (nr + 512) * Kih + kof), aZ[j]);
      }
    }
    for (int kt = 0; kt < 16; ++kt) {
      v16h a = ld_frag(hB + m * HID + kt * 32 + kh * 8);
      const int kof = kt * 32 + kh * 8;
#pragma unroll
      for (int j = 0; j < 4; ++j) {
        const size_t nr = (size_t)(ncol0 + j * 16 + m);
        aR[j] = wmma_f16(a, ld_frag(Whh + nr * HID + kof), aR[j]);
        aZ[j] = wmma_f16(a, ld_frag(Whh + (nr + 512) * HID + kof), aZ[j]);
      }
    }
#pragma unroll
    for (int j = 0; j < 4; ++j) {
      const int col = ncol0 + j * 16 + m;
      const float br = bih[col] + bhh[col];
      const float bz = bih[col + 512] + bhh[col + 512];
#pragma unroll
      for (int e = 0; e < 8; ++e) {
        const int mm = kh * 8 + e;
        rz[mm * HID + col]            = sigm(aR[j][e] + br);   // r
        rz[16 * HID + mm * HID + col] = sigm(aZ[j][e] + bz);   // z
      }
    }
  }

  // ---- pass B: n gate (input and hidden parts kept separate for r*(hn)) ----
  {
    v8f aNi[4] = { {}, {}, {}, {} };
    v8f aNh[4] = { {}, {}, {}, {} };
    for (int kt = 0; kt < K1t; ++kt) {
      v16h a = ld_frag(A1 + m * ld1 + kt * 32 + kh * 8);
      const int kof = kt * 32 + kh * 8;
#pragma unroll
      for (int j = 0; j < 4; ++j) {
        const size_t nr = (size_t)(ncol0 + j * 16 + m + 1024);
        aNi[j] = wmma_f16(a, ld_frag(Wih + nr * Kih + kof), aNi[j]);
      }
    }
    for (int kt = 0; kt < 16; ++kt) {
      v16h a = ld_frag(hB + m * HID + kt * 32 + kh * 8);
      const int kof = kt * 32 + kh * 8;
#pragma unroll
      for (int j = 0; j < 4; ++j) {
        const size_t nr = (size_t)(ncol0 + j * 16 + m + 1024);
        aNh[j] = wmma_f16(a, ld_frag(Whh + nr * HID + kof), aNh[j]);
      }
    }
    __syncthreads();  // all waves' GEMM reads of hB complete before in-place update
#pragma unroll
    for (int j = 0; j < 4; ++j) {
      const int col = ncol0 + j * 16 + m;
      const float bni = bih[col + 1024];
      const float bnh = bhh[col + 1024];
#pragma unroll
      for (int e = 0; e < 8; ++e) {
        const int mm = kh * 8 + e;                    // D-matrix row mapping
        const float r = rz[mm * HID + col];
        const float z = rz[16 * HID + mm * HID + col];
        const float n = tanhf(aNi[j][e] + bni + r * (aNh[j][e] + bnh));
        const float hold = hF[mm * HID + col];
        const float hnew = (1.f - z) * n + z * hold;
        hF[mm * HID + col] = hnew;
        hB[mm * HID + col] = (_Float16)hnew;
      }
    }
    __syncthreads();
  }
}

// ---------------- Persistent decoder: 32 WGs x 16 batch rows, 64 recurrent steps ----------------
__global__ __launch_bounds__(256)
void decoder_kernel(const _Float16* __restrict__ inph,
                    const float* __restrict__ skel,
                    const float* __restrict__ skel_enc,
                    const _Float16* __restrict__ hfc_h, const float* __restrict__ hfc_b,
                    const _Float16* __restrict__ w1ih, const _Float16* __restrict__ w1hh,
                    const float* __restrict__ b1ih, const float* __restrict__ b1hh,
                    const _Float16* __restrict__ w2ih, const _Float16* __restrict__ w2hh,
                    const float* __restrict__ b2ih, const float* __restrict__ b2hh,
                    const _Float16* __restrict__ nfc_h, const float* __restrict__ nfc_b,
                    float* __restrict__ out) {
  extern __shared__ char smem[];
  _Float16* catb = (_Float16*)smem;            // [16][288] fp16
  _Float16* h1b  = catb + 16 * K1PAD;          // [16][512] fp16
  _Float16* h2b  = h1b + 16 * HID;             // [16][512] fp16
  _Float16* outb = h2b + 16 * HID;             // [16][9] fp16 feedback (padded to 256)
  float* h1f = (float*)(outb + 256);           // [16][512] fp32
  float* h2f = h1f + 16 * HID;                 // [16][512] fp32
  float* rz  = h2f + 16 * HID;                 // [2][16][512] fp32 gate scratch

  const int tid  = threadIdx.x;
  const int lane = tid & 31;
  const int m    = lane & 15;
  const int kh   = lane >> 4;
  const int wave = tid >> 5;
  const int ncol0 = wave * 64;                 // wave's hidden-column base
  const int b0   = blockIdx.x * 16;

  // ---- init: zero cat (keeps K-pad zero forever) + feedback; stage skel_encoded as A ----
  for (int i = tid; i < 16 * K1PAD; i += 256) catb[i] = (_Float16)0.f;
  outb[tid] = (_Float16)0.f;
  for (int i = tid; i < 16 * HID; i += 256)
    h2b[i] = (_Float16)skel_enc[(size_t)(b0 + (i >> 9)) * HID + (i & 511)];
  __syncthreads();

  // ---- h1_0 = skel_enc @ hfc_w^T + hfc_b (WMMA) ----
  {
    v8f acc[4] = { {}, {}, {}, {} };
    for (int kt = 0; kt < 16; ++kt) {
      v16h a = ld_frag(h2b + m * HID + kt * 32 + kh * 8);
#pragma unroll
      for (int j = 0; j < 4; ++j)
        acc[j] = wmma_f16(a, ld_frag(hfc_h + (size_t)(ncol0 + j * 16 + m) * HID + kt * 32 + kh * 8), acc[j]);
    }
    __syncthreads();
#pragma unroll
    for (int j = 0; j < 4; ++j) {
      const int col = ncol0 + j * 16 + m;
      const float bb = hfc_b[col];
#pragma unroll
      for (int e = 0; e < 8; ++e) {
        const int mm = kh * 8 + e;
        const float v = acc[j][e] + bb;
        h1f[mm * HID + col] = v;
        h1b[mm * HID + col] = (_Float16)v;
      }
    }
    __syncthreads();
  }

  // ---- recurrent loop ----
  for (int t = 0; t < SLEN; ++t) {
    // build cat = [out(9), conv_inputs(256), skel(9)], pad already zero
    for (int i = tid; i < 16 * (H2C + NC); i += 256) {
      const int mm = i / (H2C + NC), k = i - mm * (H2C + NC);
      _Float16 v;
      if (k < H2C) v = inph[((size_t)(b0 + mm) * SLEN + t) * H2C + k];
      else         v = (_Float16)skel[((size_t)(b0 + mm) * SLEN + t) * NC + (k - H2C)];
      catb[mm * K1PAD + NC + k] = v;
    }
    for (int i = tid; i < 16 * NC; i += 256)
      catb[(i / NC) * K1PAD + (i % NC)] = outb[i];
    __syncthreads();

    gru_core(catb, 9, h1b, h1f, rz, w1ih, K1PAD, w1hh, b1ih, b1hh, m, kh, ncol0);

    if (t == 0) {  // reference: hx[1] = hx[0] at first step (after GRU1)
      for (int i = tid; i < 16 * HID; i += 256) { h2f[i] = h1f[i]; h2b[i] = h1b[i]; }
      __syncthreads();
    }

    gru_core(h1b, 16, h2b, h2f, rz, w2ih, HID, w2hh, b2ih, b2hh, m, kh, ncol0);

    // ---- logits: one 16x16x512 WMMA tile on wave 0 (batch x class), then threefry sampling ----
    if (wave == 0) {
      v8f acc = { };
      for (int kt = 0; kt < 16; ++kt) {
        v16h a = ld_frag(h2b + m * HID + kt * 32 + kh * 8);
        acc = wmma_f16(a, ld_frag(nfc_h + (size_t)m * HID + kt * 32 + kh * 8), acc);
      }
      const int j = m;                     // D-fragment: N = lane&15 = class index
      if (j < NC) {
        const float bb = nfc_b[j];
        // keys = split(key(42), 64)[t]
        unsigned k0t, k1t, a0, a1, c0, c1;
        if (t < 32) {
          tf2x32(0u, 42u, (unsigned)(2 * t),     (unsigned)(64 + 2 * t), a0, a1); k0t = a0;
          tf2x32(0u, 42u, (unsigned)(2 * t + 1), (unsigned)(65 + 2 * t), c0, c1); k1t = c0;
        } else {
          tf2x32(0u, 42u, (unsigned)(2 * t - 64), (unsigned)(2 * t),     a0, a1); k0t = a1;
          tf2x32(0u, 42u, (unsigned)(2 * t - 63), (unsigned)(2 * t + 1), c0, c1); k1t = c1;
        }
#pragma unroll
        for (int e = 0; e < 8; ++e) {
          const int mm = kh * 8 + e;
          const float lg = acc[e] + bb;
          out[((size_t)(b0 + mm) * SLEN + t) * NC + j] = lg;
          const unsigned idx = (unsigned)((b0 + mm) * NC + j);   // element of (512,9) uniform draw
          unsigned bits;
          if (idx < 2304u) { tf2x32(k0t, k1t, idx, 2304u + idx, a0, a1); bits = a0; }
          else             { tf2x32(k0t, k1t, idx - 2304u, idx, a0, a1); bits = a1; }
          const float u = __uint_as_float(0x3f800000u | (bits >> 9)) - 1.0f;
          const float s = 1.f / (1.f + expf(-lg));
          outb[mm * NC + j] = (s - u > 0.f) ? (_Float16)1.f : (_Float16)0.f;
        }
      }
    }
    __syncthreads();
  }
}

// ---------------- host launch ----------------
extern "C" void kernel_launch(void* const* d_in, const int* in_sizes, int n_in,
                              void* d_out, int out_size, void* d_ws, size_t ws_size,
                              hipStream_t stream) {
  const float* z        = (const float*)d_in[0];
  const float* skel     = (const float*)d_in[2];
  const float* skel_enc = (const float*)d_in[3];
  const float* ct1w = (const float*)d_in[6];
  const float* ct1b = (const float*)d_in[7];
  const float* bn1g = (const float*)d_in[8];
  const float* bn1b = (const float*)d_in[9];
  const float* bn1m = (const float*)d_in[10];
  const float* bn1v = (const float*)d_in[11];
  const float* ct2w = (const float*)d_in[12];
  const float* ct2b = (const float*)d_in[13];
  const float* bn2g = (const float*)d_in[14];
  const float* bn2b = (const float*)d_in[15];
  const float* bn2m = (const float*)d_in[16];
  const float* bn2v = (const float*)d_in[17];
  const float* hfcw = (const float*)d_in[18];
  const float* hfcb = (const float*)d_in[19];
  const float* g1wih = (const float*)d_in[20];
  const float* g1whh = (const float*)d_in[21];
  const float* g1bih = (const float*)d_in[22];
  const float* g1bhh = (const float*)d_in[23];
  const float* g2wih = (const float*)d_in[24];
  const float* g2whh = (const float*)d_in[25];
  const float* g2bih = (const float*)d_in[26];
  const float* g2bhh = (const float*)d_in[27];
  const float* nfcw  = (const float*)d_in[28];
  const float* nfcb  = (const float*)d_in[29];
  float* out = (float*)d_out;

  char* ws = (char*)d_ws;
  size_t off = 0;
  float*    y1   = (float*)(ws + off);    off += (size_t)512 * 32 * 512 * 4;   // conv1 out [B][32][512]
  _Float16* inph = (_Float16*)(ws + off); off += (size_t)512 * 64 * 256 * 2;   // conv2 out fp16
  _Float16* w1ih = (_Float16*)(ws + off); off += (size_t)1536 * 288 * 2;
  _Float16* w1hh = (_Float16*)(ws + off); off += (size_t)1536 * 512 * 2;
  _Float16* w2ih = (_Float16*)(ws + off); off += (size_t)1536 * 512 * 2;
  _Float16* w2hh = (_Float16*)(ws + off); off += (size_t)1536 * 512 * 2;
  _Float16* hfch = (_Float16*)(ws + off); off += (size_t)512 * 512 * 2;
  _Float16* nfch = (_Float16*)(ws + off); off += (size_t)16 * 512 * 2;
  (void)ws_size; (void)in_sizes; (void)n_in; (void)out_size;

  auto nb = [](int n) { return (n + 255) / 256; };
  cvt_pad_half<<<nb(1536 * 288), 256, 0, stream>>>(g1wih, w1ih, 1536, 1536, 274, 288);
  cvt_pad_half<<<nb(1536 * 512), 256, 0, stream>>>(g1whh, w1hh, 1536, 1536, 512, 512);
  cvt_pad_half<<<nb(1536 * 512), 256, 0, stream>>>(g2wih, w2ih, 1536, 1536, 512, 512);
  cvt_pad_half<<<nb(1536 * 512), 256, 0, stream>>>(g2whh, w2hh, 1536, 1536, 512, 512);
  cvt_pad_half<<<nb(512 * 512),  256, 0, stream>>>(hfcw,  hfch, 512,  512,  512, 512);
  cvt_pad_half<<<nb(16 * 512),   256, 0, stream>>>(nfcw,  nfch, 9,    16,   512, 512);

  convt1_kernel<<<512 * 32, 512, 0, stream>>>(z, ct1w, ct1b, bn1g, bn1b, bn1m, bn1v, y1);
  convt2_kernel<<<512 * 64, 256, 0, stream>>>(y1, ct2w, ct2b, bn2g, bn2b, bn2m, bn2v, inph);

  const size_t smem = (size_t)16 * K1PAD * 2      // cat fp16
                    + (size_t)16 * 512 * 2 * 2    // h1b, h2b fp16
                    + 256 * 2                     // outb
                    + (size_t)16 * 512 * 4 * 2    // h1f, h2f fp32
                    + (size_t)16 * 512 * 4 * 2;   // rz gate scratch fp32  (~170 KB < 320 KB/WGP)
  decoder_kernel<<<32, 256, smem, stream>>>(inph, skel, skel_enc, hfch, hfcb,
                                            w1ih, w1hh, g1bih, g1bhh,
                                            w2ih, w2hh, g2bih, g2bhh,
                                            nfch, nfcb, out);
}
